// mRNN_4002909520796
// MI455X (gfx1250) — matrix operand order repaired and checked by reading, not compiled, not run on hardware
//
#include <hip/hip_runtime.h>

typedef __attribute__((ext_vector_type(16))) __bf16 v16bf;
typedef __attribute__((ext_vector_type(8)))  float  v8f;

#define N_  1024
#define I_  128
#define B_  64
#define T_  300
#define K_  (N_ + I_)        // 1152 concatenated reduction dim
#define LDSROW 1160          // padded bf16 row: 2320B -> 16 fragment rows hit distinct banks
#define ALPHA 0.1f
#define KREC  0.0044721359549995794f   // sqrt(2*0.1*0.01^2)
#define KINP  0.0044721359549995794f

union FragBF { v16bf v; uint4 u4[2]; };
union FragF  { v8f  v; float f[8];  };

// ---- one-time: effective weights -> bf16 concat panel [N][K] --------------
__global__ __launch_bounds__(256) void prep_weights(
    const float* __restrict__ Wr,  const float* __restrict__ Wi,
    const float* __restrict__ Wrm, const float* __restrict__ Wrs,
    const float* __restrict__ Wim, const float* __restrict__ Wis,
    __bf16* __restrict__ Wcat)
{
  int idx = blockIdx.x * blockDim.x + threadIdx.x;
  if (idx >= N_ * K_) return;
  int n = idx / K_, k = idx - n * K_;
  float v;
  if (k < N_) {
    float w = fminf(fmaxf(Wr[n * N_ + k], 0.f), 10.f);
    v = Wrm[n * N_ + k] * Wrs[n * N_ + k] * w;
  } else {
    int i = k - N_;
    float w = fminf(fmaxf(Wi[n * I_ + i], 0.f), 10.f);
    v = Wim[n * I_ + i] * Wis[n * I_ + i] * w;
  }
  Wcat[idx] = (__bf16)v;
}

// ---- persistent scan: 1 workgroup per 16-row batch block ------------------
// 1024 threads = 32 waves; wave w owns output columns [w*32, w*32+32) (2 WMMA tiles).
__global__ __launch_bounds__(1024) void rnn_persistent(
    const float*  __restrict__ inp,      // [B,T,I]
    const float*  __restrict__ x0,       // [B,N]
    const __bf16* __restrict__ Wcat,     // [N,K] bf16
    const float*  __restrict__ baseline, // [N]
    const float*  __restrict__ eps_rec,  // [T,B,N]
    const float*  __restrict__ eps_inp,  // [T,B,I]
    float*        __restrict__ out)      // [B,T,N]
{
  __shared__ alignas(16) __bf16 hb[16 * LDSROW];   // [16 rows][1024 h | 128 u_noisy]

  const int tid  = threadIdx.x;
  const int lane = tid & 31;
  const int wave = tid >> 5;            // 0..31
  const int b0   = blockIdx.x * 16;     // batch rows of this block
  const int n0   = wave * 32;           // first of 2 column tiles
  const int l16  = lane & 15;
  const int half = lane >> 4;           // 0|1
  const int urow = tid >> 6;            // 0..15  (u_noisy fill mapping)
  const int uc0  = (tid & 63) * 2;      // 0..126

  // prologue: h(=x0) and u_noisy(t=0) into LDS (bf16)
  for (int e = tid; e < 16 * K_; e += 1024) {
    int m = e / K_, k = e - m * K_;
    float v;
    if (k < N_) {
      v = x0[(size_t)(b0 + m) * N_ + k];
    } else {
      int c = k - N_;
      v = inp[((size_t)(b0 + m) * T_ + 0) * I_ + c]
        + KINP * eps_inp[(size_t)(b0 + m) * I_ + c];
    }
    hb[(size_t)m * LDSROW + k] = (__bf16)v;
  }

  // f32 master h in registers, mapped exactly like the WMMA C tile
  FragF h[2];
  float bl[2];
  #pragma unroll
  for (int tl = 0; tl < 2; ++tl) {
    int n = n0 + tl * 16 + l16;
    bl[tl] = baseline[n];
    #pragma unroll
    for (int r = 0; r < 8; ++r) {
      int m = r + half * 8;
      h[tl].f[r] = x0[(size_t)(b0 + m) * N_ + n];
    }
  }
  __syncthreads();

  const int m = l16;  // A-fragment row for this lane
  for (int t = 0; t < T_; ++t) {
    FragF acc[2];
    #pragma unroll
    for (int tl = 0; tl < 2; ++tl)
      #pragma unroll
      for (int r = 0; r < 8; ++r) acc[tl].f[r] = 0.0f;

    // K loop: 36 chunks of 32 (bf16 WMMA), weights streamed from L2-resident panel.
    // LOADcnt stays dedicated to B-fragment loads here -> scheduler keeps slack waits.
    #pragma unroll 4
    for (int kc = 0; kc < K_; kc += 32) {
      FragBF a;
      const __bf16* ap = hb + (size_t)m * LDSROW + kc + half * 8;
      a.u4[0] = *(const uint4*)(ap);
      a.u4[1] = *(const uint4*)(ap + 16);
      #pragma unroll
      for (int tl = 0; tl < 2; ++tl) {
        FragBF b;
        const __bf16* bp = Wcat + (size_t)(n0 + tl * 16 + l16) * K_ + kc + half * 8;
        b.u4[0] = *(const uint4*)(bp);
        b.u4[1] = *(const uint4*)(bp + 16);
        acc[tl].v = __builtin_amdgcn_wmma_f32_16x16x32_bf16(
            false, a.v, false, b.v, (short)0, acc[tl].v, false, false);
      }
    }

    // ---- issue streaming loads now: they overlap the barrier and the other
    //      waves' K-loop tails, and are consumed only after the barrier.
    float er_pre[2][8];
    #pragma unroll
    for (int tl = 0; tl < 2; ++tl) {
      int n = n0 + tl * 16 + l16;
      #pragma unroll
      for (int r = 0; r < 8; ++r) {
        int b = b0 + r + half * 8;
        er_pre[tl][r] =
            __builtin_nontemporal_load(&eps_rec[((size_t)t * B_ + b) * N_ + n]);
      }
    }
    float u_pre[2];
    if (t + 1 < T_) {
      int b = b0 + urow;
      #pragma unroll
      for (int j = 0; j < 2; ++j) {
        int c = uc0 + j;
        u_pre[j] = inp[((size_t)b * T_ + (t + 1)) * I_ + c]
                 + KINP * __builtin_nontemporal_load(
                       &eps_inp[((size_t)(t + 1) * B_ + b) * I_ + c]);
      }
    }

    __syncthreads();   // all waves done reading h/u from LDS

    // epilogue: noise + baseline, tau blend, write out / LDS (owned cells only)
    #pragma unroll
    for (int tl = 0; tl < 2; ++tl) {
      int n = n0 + tl * 16 + l16;
      #pragma unroll
      for (int r = 0; r < 8; ++r) {
        int mm = r + half * 8;
        int b  = b0 + mm;
        float drive = acc[tl].f[r] + bl[tl] + KREC * er_pre[tl][r];
        float hn    = (1.0f - ALPHA) * h[tl].f[r] + ALPHA * fmaxf(drive, 0.0f);
        h[tl].f[r] = hn;
        __builtin_nontemporal_store(hn, &out[((size_t)b * T_ + t) * N_ + n]);
        hb[(size_t)mm * LDSROW + n] = (__bf16)hn;
      }
    }
    // next step's noisy input -> LDS tail (loaded early above)
    if (t + 1 < T_) {
      #pragma unroll
      for (int j = 0; j < 2; ++j) {
        hb[(size_t)urow * LDSROW + N_ + uc0 + j] = (__bf16)u_pre[j];
      }
    }
    __syncthreads();   // writes visible before next step reads
  }
}

extern "C" void kernel_launch(void* const* d_in, const int* in_sizes, int n_in,
                              void* d_out, int out_size, void* d_ws, size_t ws_size,
                              hipStream_t stream) {
  const float* inp  = (const float*)d_in[0];
  const float* x0   = (const float*)d_in[1];
  const float* Wr   = (const float*)d_in[2];
  const float* Wi   = (const float*)d_in[3];
  const float* Wrm  = (const float*)d_in[4];
  const float* Wrs  = (const float*)d_in[5];
  const float* Wim  = (const float*)d_in[6];
  const float* Wis  = (const float*)d_in[7];
  const float* base = (const float*)d_in[8];
  const float* er   = (const float*)d_in[9];
  const float* ei   = (const float*)d_in[10];

  __bf16* Wcat = (__bf16*)d_ws;   // needs N*K*2 = 2.25 MB of workspace

  int total = N_ * K_;
  prep_weights<<<(total + 255) / 256, 256, 0, stream>>>(Wr, Wi, Wrm, Wrs, Wim, Wis, Wcat);
  rnn_persistent<<<4, 1024, 0, stream>>>(inp, x0, Wcat, base, er, ei, (float*)d_out);
}
